// LongcatFlashTopkRouter_37752762532031
// MI455X (gfx1250) — matrix operand order repaired and checked by reading, not compiled, not run on hardware
//
#include <hip/hip_runtime.h>
#include <math.h>

typedef __attribute__((ext_vector_type(16))) __bf16       v16bf;
typedef __attribute__((ext_vector_type(8)))  float        v8f;
typedef __attribute__((ext_vector_type(4)))  unsigned int v4u;

#define HID   4096
#define NEXP  256
#define TOPK  8
#define KT    32      // K per LDS stage == WMMA K
#define TTILE 128     // tokens per workgroup
#define ETILE 64      // experts per workgroup

union FragAB {
    v4u   u[2];   // two 16-byte chunks
    v16bf b;      // 16 bf16 -> 8 VGPRs, matches WMMA A/B fragment
};

// Split one float4 into bf16 hi/lo pairs and store 8 bytes to each LDS tile.
__device__ __forceinline__ void cvt_store8(__bf16* hiP, __bf16* loP, float4 v) {
    union { __bf16 h[4]; unsigned long long u; } H, L;
    float f0 = v.x, f1 = v.y, f2 = v.z, f3 = v.w;
    __bf16 h0 = (__bf16)f0, h1 = (__bf16)f1, h2 = (__bf16)f2, h3 = (__bf16)f3;
    H.h[0] = h0; H.h[1] = h1; H.h[2] = h2; H.h[3] = h3;
    L.h[0] = (__bf16)(f0 - (float)h0);
    L.h[1] = (__bf16)(f1 - (float)h1);
    L.h[2] = (__bf16)(f2 - (float)h2);
    L.h[3] = (__bf16)(f3 - (float)h3);
    *(unsigned long long*)hiP = H.u;
    *(unsigned long long*)loP = L.u;
}

// ---------------------------------------------------------------------------
// Kernel 1: classify = X @ W^T via split-fp32 (bf16x3) WMMA.
// X: [T, 4096] f32 row-major, W: [256, 4096] f32 row-major, C: [T, 256] f32.
// ---------------------------------------------------------------------------
__global__ __launch_bounds__(256)
void router_gemm_bf16x3(const float* __restrict__ X, const float* __restrict__ W,
                        float* __restrict__ C) {
    __shared__ __bf16 sAh[2][TTILE][KT];
    __shared__ __bf16 sAl[2][TTILE][KT];
    __shared__ __bf16 sBh[2][ETILE][KT];
    __shared__ __bf16 sBl[2][ETILE][KT];

    const int tid  = threadIdx.x;
    const int lane = tid & 31;
    const int wave = tid >> 5;        // 0..7
    const int wm   = wave >> 1;       // 0..3 : 32-token block
    const int wn   = wave & 1;        // 0..1 : 32-expert block
    const int lm   = lane & 15;
    const int lh   = lane >> 4;

    const int tRow0 = blockIdx.y * TTILE;
    const int eCol0 = blockIdx.x * ETILE;

    float4 aReg[4];   // 128*32 f32 / 256 threads = 16 elems
    float4 bReg[2];   //  64*32 f32 / 256 threads =  8 elems

    auto gload = [&](int k0) {
#pragma unroll
        for (int i = 0; i < 4; ++i) {
            int f = tid + i * 256;            // float4 index 0..1023
            int r = f >> 3, c = (f & 7) << 2; // row 0..127, col 0..28
            aReg[i] = *(const float4*)(X + (size_t)(tRow0 + r) * HID + k0 + c);
        }
#pragma unroll
        for (int i = 0; i < 2; ++i) {
            int f = tid + i * 256;            // 0..511
            int r = f >> 3, c = (f & 7) << 2; // row 0..63
            bReg[i] = *(const float4*)(W + (size_t)(eCol0 + r) * HID + k0 + c);
        }
    };

    auto lstore = [&](int buf) {
#pragma unroll
        for (int i = 0; i < 4; ++i) {
            int f = tid + i * 256;
            int r = f >> 3, c = (f & 7) << 2;
            cvt_store8(&sAh[buf][r][c], &sAl[buf][r][c], aReg[i]);
        }
#pragma unroll
        for (int i = 0; i < 2; ++i) {
            int f = tid + i * 256;
            int r = f >> 3, c = (f & 7) << 2;
            cvt_store8(&sBh[buf][r][c], &sBl[buf][r][c], bReg[i]);
        }
    };

    v8f acc[2][2] = {};   // 32x32 per wave = 2x2 WMMA fragments

    auto compute = [&](int buf) {
        FragAB ah[2], al[2], bh[2], bl[2];
        // A fragments: lane lm = row, chunks K[lh*8..+7] and K[16+lh*8..+7]
#pragma unroll
        for (int rb = 0; rb < 2; ++rb) {
            int r = wm * 32 + rb * 16 + lm;
            ah[rb].u[0] = *(const v4u*)&sAh[buf][r][lh * 8];
            ah[rb].u[1] = *(const v4u*)&sAh[buf][r][16 + lh * 8];
            al[rb].u[0] = *(const v4u*)&sAl[buf][r][lh * 8];
            al[rb].u[1] = *(const v4u*)&sAl[buf][r][16 + lh * 8];
        }
        // B fragments: lane lm = column (W row), chunks K[lh*8..+7] and K[16+lh*8..+7]
#pragma unroll
        for (int nb = 0; nb < 2; ++nb) {
            int e = wn * 32 + nb * 16 + lm;
            bh[nb].u[0] = *(const v4u*)&sBh[buf][e][lh * 8];
            bh[nb].u[1] = *(const v4u*)&sBh[buf][e][16 + lh * 8];
            bl[nb].u[0] = *(const v4u*)&sBl[buf][e][lh * 8];
            bl[nb].u[1] = *(const v4u*)&sBl[buf][e][16 + lh * 8];
        }
#pragma unroll
        for (int rb = 0; rb < 2; ++rb)
#pragma unroll
            for (int nb = 0; nb < 2; ++nb) {
                acc[rb][nb] = __builtin_amdgcn_wmma_f32_16x16x32_bf16(
                    false, ah[rb].b, false, bh[nb].b, (short)0, acc[rb][nb], false, false);
                acc[rb][nb] = __builtin_amdgcn_wmma_f32_16x16x32_bf16(
                    false, ah[rb].b, false, bl[nb].b, (short)0, acc[rb][nb], false, false);
                acc[rb][nb] = __builtin_amdgcn_wmma_f32_16x16x32_bf16(
                    false, al[rb].b, false, bh[nb].b, (short)0, acc[rb][nb], false, false);
            }
    };

    const int KITERS = HID / KT;   // 128
    gload(0);
    lstore(0);
    __syncthreads();

    for (int it = 0; it < KITERS; ++it) {
        const int buf = it & 1;
        if (it + 1 < KITERS) gload((it + 1) * KT);   // prefetch next tile (global)
        compute(buf);                                 // WMMA on current tile
        __syncthreads();                              // everyone done reading buf^1's prior use
        if (it + 1 < KITERS) lstore(buf ^ 1);         // convert+stage next tile
        __syncthreads();                              // staged tile visible before next compute
    }

    // Epilogue: C layout — lane lm = column, rows lh*8 + i per fragment.
#pragma unroll
    for (int rb = 0; rb < 2; ++rb)
#pragma unroll
        for (int nb = 0; nb < 2; ++nb) {
            int col  = eCol0 + wn * 32 + nb * 16 + lm;
            int row0 = tRow0 + wm * 32 + rb * 16 + lh * 8;
#pragma unroll
            for (int i = 0; i < 8; ++i)
                C[(size_t)(row0 + i) * NEXP + col] = acc[rb][nb][i];
        }
}

// ---------------------------------------------------------------------------
// Kernel 2: per-token softmax, bias-corrected top-8, gather unbiased * 1.5.
// One wave per token; lane holds 8 contiguous experts.
// ---------------------------------------------------------------------------
__global__ __launch_bounds__(256)
void router_topk(const float* __restrict__ S, const float* __restrict__ bias,
                 float* __restrict__ wOut, int* __restrict__ iOut) {
    const int t    = blockIdx.x * 8 + (threadIdx.x >> 5);
    const int lane = threadIdx.x & 31;
    const float* row = S + (size_t)t * NEXP;

    float v[8];
    float4 x0 = *(const float4*)(row + lane * 8);
    float4 x1 = *(const float4*)(row + lane * 8 + 4);
    v[0] = x0.x; v[1] = x0.y; v[2] = x0.z; v[3] = x0.w;
    v[4] = x1.x; v[5] = x1.y; v[6] = x1.z; v[7] = x1.w;

    float m = v[0];
#pragma unroll
    for (int j = 1; j < 8; ++j) m = fmaxf(m, v[j]);
#pragma unroll
    for (int off = 16; off > 0; off >>= 1) m = fmaxf(m, __shfl_xor(m, off, 32));

    float pr[8];
    float s = 0.f;
#pragma unroll
    for (int j = 0; j < 8; ++j) { pr[j] = __expf(v[j] - m); s += pr[j]; }
#pragma unroll
    for (int off = 16; off > 0; off >>= 1) s += __shfl_xor(s, off, 32);
    float inv = 1.0f / s;
#pragma unroll
    for (int j = 0; j < 8; ++j) pr[j] *= inv;

    float4 b0 = *(const float4*)(bias + lane * 8);
    float4 b1 = *(const float4*)(bias + lane * 8 + 4);
    float bs[8];
    bs[0] = pr[0] + b0.x; bs[1] = pr[1] + b0.y; bs[2] = pr[2] + b0.z; bs[3] = pr[3] + b0.w;
    bs[4] = pr[4] + b1.x; bs[5] = pr[5] + b1.y; bs[6] = pr[6] + b1.z; bs[7] = pr[7] + b1.w;

    for (int k = 0; k < TOPK; ++k) {
        // lane-local argmax (ties -> lowest index)
        float bestV = bs[0];
        int   bestJ = 0;
#pragma unroll
        for (int j = 1; j < 8; ++j)
            if (bs[j] > bestV) { bestV = bs[j]; bestJ = j; }
        int bestE = lane * 8 + bestJ;
        // wave argmax (ties -> lowest expert index)
#pragma unroll
        for (int off = 16; off > 0; off >>= 1) {
            float ov = __shfl_xor(bestV, off, 32);
            int   oe = __shfl_xor(bestE, off, 32);
            if (ov > bestV || (ov == bestV && oe < bestE)) { bestV = ov; bestE = oe; }
        }
        // gather UNbiased prob of the winner
        float sel = 0.f;
#pragma unroll
        for (int j = 0; j < 8; ++j)
            if ((bestE & 7) == j) sel = pr[j];
        float w = __shfl(sel, bestE >> 3, 32);

        if (lane == 0) {
            wOut[(size_t)t * TOPK + k] = w * 1.5f;
            iOut[(size_t)t * TOPK + k] = bestE;
        }
        if ((bestE >> 3) == lane) bs[bestE & 7] = -3.402823466e+38f;
    }
}

// ---------------------------------------------------------------------------
extern "C" void kernel_launch(void* const* d_in, const int* in_sizes, int n_in,
                              void* d_out, int out_size, void* d_ws, size_t ws_size,
                              hipStream_t stream) {
    (void)n_in; (void)out_size; (void)d_ws; (void)ws_size;
    const float* hidden = (const float*)d_in[0];   // [T, 4096] f32
    const float* W      = (const float*)d_in[1];   // [256, 4096] f32
    const float* bias   = (const float*)d_in[2];   // [256] f32

    const int T = in_sizes[0] / HID;               // 16384

    float* out      = (float*)d_out;
    float* classify = out;                                          // T*256
    float* weights  = out + (size_t)T * NEXP;                       // T*8
    int*   indices  = (int*)(out + (size_t)T * NEXP + (size_t)T * TOPK);

    dim3 gGemm(NEXP / ETILE, T / TTILE);           // 4 x 128
    router_gemm_bf16x3<<<gGemm, 256, 0, stream>>>(hidden, W, classify);
    router_topk<<<T / 8, 256, 0, stream>>>(classify, bias, weights, indices);
}
